// MOELayer_61933428408751
// MI455X (gfx1250) — compile-verified, matrix operand order checked by reading, not run on hardware
//
#include <hip/hip_runtime.h>

// ---------------- types ----------------
typedef __attribute__((ext_vector_type(16))) __bf16 v16bf;
typedef __attribute__((ext_vector_type(8)))  __bf16 bf16x8;
typedef __attribute__((ext_vector_type(8)))  float  v8f;

#define TOK_N   8192          // S*T tokens
#define MDIM    2048
#define EXP_N   8
#define HDIM    8192
#define CAP     1024          // capacity per expert
#define LDT     40            // LDS row stride (bf16 elems): 32 + 8 pad
#define TILE_ELEMS (128 * LDT)

// =====================================================================
// 1. init small arrays (slot_token = -1, slot_gate = 0)
// =====================================================================
__global__ __launch_bounds__(256) void init_kernel(int* slot_token, float* slot_gate) {
    int i = blockIdx.x * 256 + threadIdx.x;
    if (i < EXP_N * CAP) { slot_token[i] = -1; slot_gate[i] = 0.0f; }
}

// zero the combined output region (16777216 f32) with float4 stores
__global__ __launch_bounds__(256) void zero_out_kernel(float* out) {
    int gid = blockIdx.x * 256 + threadIdx.x;      // 16384*256 = 4194304 float4
    ((float4*)out)[gid] = make_float4(0.f, 0.f, 0.f, 0.f);
}

// =====================================================================
// 2. gating: one wave per token
// =====================================================================
__global__ __launch_bounds__(256) void gate_kernel(const float* __restrict__ x,
                                                   const float* __restrict__ wg,
                                                   int* __restrict__ idx,
                                                   float* __restrict__ gate1,
                                                   float* __restrict__ gates) {
    int lane = threadIdx.x & 31;
    int n    = blockIdx.x * 8 + (threadIdx.x >> 5);
    const float* xp = x + (size_t)n * MDIM;
    float acc[EXP_N];
#pragma unroll
    for (int e = 0; e < EXP_N; ++e) acc[e] = 0.f;
    for (int j = lane; j < MDIM; j += 32) {
        float xv = xp[j];
#pragma unroll
        for (int e = 0; e < EXP_N; ++e) acc[e] += xv * wg[e * MDIM + j];
    }
#pragma unroll
    for (int e = 0; e < EXP_N; ++e)
#pragma unroll
        for (int off = 16; off > 0; off >>= 1) acc[e] += __shfl_xor(acc[e], off, 32);
    if (lane == 0) {
        float mx = acc[0]; int am = 0;
#pragma unroll
        for (int e = 1; e < EXP_N; ++e) if (acc[e] > mx) { mx = acc[e]; am = e; }
        float g[EXP_N]; float s = 0.f;
#pragma unroll
        for (int e = 0; e < EXP_N; ++e) { g[e] = __expf(acc[e] - mx); s += g[e]; }
        float inv = 1.f / s;
#pragma unroll
        for (int e = 0; e < EXP_N; ++e) { g[e] *= inv; gates[n * EXP_N + e] = g[e]; }
        idx[n] = am;
        gate1[n] = g[am];
    }
}

// deterministic me[e] = sum_n gates[n][e]  (one block per expert)
__global__ __launch_bounds__(256) void me_reduce_kernel(const float* __restrict__ gates,
                                                        float* __restrict__ me) {
    __shared__ float sh[256];
    int e = blockIdx.x, t = threadIdx.x;
    float s = 0.f;
    for (int n = t; n < TOK_N; n += 256) s += gates[n * EXP_N + e];
    sh[t] = s; __syncthreads();
    for (int o = 128; o > 0; o >>= 1) { if (t < o) sh[t] += sh[t + o]; __syncthreads(); }
    if (t == 0) me[e] = sh[0];
}

// =====================================================================
// 3. ordered capacity assignment (cumsum semantics)
// =====================================================================
__global__ __launch_bounds__(256) void hist_kernel(const int* __restrict__ idx,
                                                   int* __restrict__ hist) {
    __shared__ int sh[EXP_N];
    int t = threadIdx.x;
    if (t < EXP_N) sh[t] = 0;
    __syncthreads();
    int e = idx[blockIdx.x * 256 + t];
    atomicAdd(&sh[e], 1);
    __syncthreads();
    if (t < EXP_N) hist[blockIdx.x * EXP_N + t] = sh[t];
}

__global__ void scan_kernel(const int* __restrict__ hist, int* __restrict__ base,
                            float* __restrict__ cef) {
    int e = threadIdx.x;
    if (e >= EXP_N) return;
    int run = 0;
    for (int b = 0; b < 32; ++b) { base[b * EXP_N + e] = run; run += hist[b * EXP_N + e]; }
    cef[e] = (float)run;
}

__global__ __launch_bounds__(256) void assign_kernel(const int* __restrict__ idx,
                                                     const float* __restrict__ gate1,
                                                     const int* __restrict__ base,
                                                     int* __restrict__ slot_token,
                                                     float* __restrict__ slot_gate) {
    __shared__ int wcnt[8][EXP_N];
    int t = threadIdx.x, lane = t & 31, w = t >> 5;
    int n = blockIdx.x * 256 + t;
    int e = idx[n];
    int myrank = 0;
#pragma unroll
    for (int j = 0; j < EXP_N; ++j) {
        unsigned m = __builtin_amdgcn_ballot_w32(e == j);   // uniform control flow
        if (lane == 0) wcnt[w][j] = __popc(m);
        if (e == j) myrank = __popc(m & ((1u << lane) - 1u));
    }
    __syncthreads();
    int wb = base[blockIdx.x * EXP_N + e];
    for (int w2 = 0; w2 < w; ++w2) wb += wcnt[w2][e];
    int loc = wb + myrank;
    if (loc < CAP) {
        int slot = e * CAP + loc;
        slot_token[slot] = n;
        slot_gate[slot]  = gate1[n];
    }
}

__global__ void laux_kernel(const float* __restrict__ me, const float* __restrict__ cef,
                            float* __restrict__ out) {
    if (threadIdx.x == 0 && blockIdx.x == 0) {
        float s = 0.f;
#pragma unroll
        for (int e = 0; e < EXP_N; ++e) s += me[e] * cef[e];
        out[(size_t)TOK_N * MDIM] = s * ((float)EXP_N / ((float)TOK_N * (float)TOK_N));
    }
}

// =====================================================================
// 4. dispatch: gather tokens -> bf16 [E*CAP, M]
// =====================================================================
__global__ __launch_bounds__(256) void dispatch_kernel(const float* __restrict__ x,
                                                       const int* __restrict__ slot_token,
                                                       __bf16* __restrict__ disp) {
    int slot = blockIdx.x;
    int tok  = slot_token[slot];
    const float* xp = x + (size_t)(tok >= 0 ? tok : 0) * MDIM;
#pragma unroll
    for (int k = 0; k < 8; ++k) {
        int c = threadIdx.x + k * 256;
        float v = (tok >= 0) ? xp[c] : 0.f;
        disp[(size_t)slot * MDIM + c] = (__bf16)v;
    }
}

// =====================================================================
// 5/6. WMMA GEMM core: block tile 128x128, BK=32, 8 waves, wave = 64x32
//      double-buffered LDS, packed-dword B staging
// =====================================================================
__device__ inline v16bf ld_frag(const __bf16* s, int rowbase, int lane) {
    int r  = rowbase + (lane & 15);
    int ko = (lane & 16) >> 1;                     // K half per lane group
    const __bf16* p = s + r * LDT + ko;
    union { v16bf v; bf16x8 h[2]; } u;
    u.h[0] = *(const bf16x8*)(p);                  // K = ko .. ko+7
    u.h[1] = *(const bf16x8*)(p + 16);             // K = ko+16 .. ko+23
    return u.v;
}

__device__ inline void gemm_core(const __bf16* __restrict__ A, int lda,
                                 const float* __restrict__ Bg, int ldb,
                                 int K, int rbase, int nbase,
                                 __bf16* As, __bf16* Bs, v8f acc[4][2]) {
    const int t    = threadIdx.x;
    const int lane = t & 31;
    const int w    = t >> 5;
    const int wm   = w & 1, wn = w >> 1;
    const int ar   = t >> 1, akc = (t & 1) << 4;   // A-tile loader coords
    const int kp   = (t & 15) << 1;                // B k-pair base: 0,2,..,30
    const int ng   = (t >> 4) << 3;                // B n-group base: 0,8,..,120

    const __bf16* aptr = A + (size_t)(rbase + ar) * lda + akc;
    const float*  bptr = Bg + (size_t)kp * ldb + nbase + ng;

    bf16x8 a0, a1;
    float4 f0, f1, f2, f3;

    // prologue: stage tile k0 = 0 into registers
    a0 = *(const bf16x8*)(aptr);
    a1 = *(const bf16x8*)(aptr + 8);
    f0 = *(const float4*)(bptr);
    f1 = *(const float4*)(bptr + 4);
    f2 = *(const float4*)(bptr + ldb);
    f3 = *(const float4*)(bptr + ldb + 4);

    int cur = 0;
    for (int k0 = 0; k0 < K; k0 += 32) {
        __bf16* Ab = As + cur * TILE_ELEMS;
        __bf16* Bb = Bs + cur * TILE_ELEMS;

        // commit staged registers to LDS buffer `cur`
        *(bf16x8*)&Ab[ar * LDT + akc]     = a0;
        *(bf16x8*)&Ab[ar * LDT + akc + 8] = a1;
        {
            float r0[8] = { f0.x, f0.y, f0.z, f0.w, f1.x, f1.y, f1.z, f1.w };
            float r1[8] = { f2.x, f2.y, f2.z, f2.w, f3.x, f3.y, f3.z, f3.w };
#pragma unroll
            for (int i = 0; i < 8; ++i) {
                union { unsigned u; __bf16 b[2]; } p;
                p.b[0] = (__bf16)r0[i];            // K = kp
                p.b[1] = (__bf16)r1[i];            // K = kp+1 (fragment dword pair)
                *(unsigned*)&Bb[(ng + i) * LDT + kp] = p.u;
            }
        }
        __syncthreads();

        // issue next tile's global loads early (overlap with WMMA below)
        if (k0 + 32 < K) {
            aptr += 32;
            bptr += (size_t)32 * ldb;
            a0 = *(const bf16x8*)(aptr);
            a1 = *(const bf16x8*)(aptr + 8);
            f0 = *(const float4*)(bptr);
            f1 = *(const float4*)(bptr + 4);
            f2 = *(const float4*)(bptr + ldb);
            f3 = *(const float4*)(bptr + ldb + 4);
        }

        v16bf af[4], bfr[2];
#pragma unroll
        for (int i = 0; i < 4; ++i) af[i]  = ld_frag(Ab, wm * 64 + i * 16, lane);
#pragma unroll
        for (int j = 0; j < 2; ++j) bfr[j] = ld_frag(Bb, wn * 32 + j * 16, lane);
#pragma unroll
        for (int i = 0; i < 4; ++i)
#pragma unroll
            for (int j = 0; j < 2; ++j)
                acc[i][j] = __builtin_amdgcn_wmma_f32_16x16x32_bf16(
                    false, af[i], false, bfr[j], (short)0, acc[i][j], false, false);

        cur ^= 1;
    }
}

// GEMM1: h = relu(disp @ w1 + b1)    [per expert: 1024x2048 @ 2048x8192]
__global__ __launch_bounds__(256) void gemm1_kernel(const __bf16* __restrict__ disp,
                                                    const float* __restrict__ w1,
                                                    const float* __restrict__ b1,
                                                    __bf16* __restrict__ h) {
    __shared__ __bf16 As[2 * TILE_ELEMS];
    __shared__ __bf16 Bs[2 * TILE_ELEMS];
    int e = blockIdx.z;
    int rbase = blockIdx.y * 128, nbase = blockIdx.x * 128;
    v8f acc[4][2];
#pragma unroll
    for (int i = 0; i < 4; ++i)
#pragma unroll
        for (int j = 0; j < 2; ++j)
#pragma unroll
            for (int v = 0; v < 8; ++v) acc[i][j][v] = 0.f;

    gemm_core(disp + (size_t)e * CAP * MDIM, MDIM,
              w1 + (size_t)e * MDIM * HDIM, HDIM,
              MDIM, rbase, nbase, As, Bs, acc);

    int lane = threadIdx.x & 31, w = threadIdx.x >> 5;
    int wm = w & 1, wn = w >> 1;
    int nlo = lane & 15, rhi = (lane & 16) >> 1;   // +8 rows for upper lane half
#pragma unroll
    for (int i = 0; i < 4; ++i)
#pragma unroll
        for (int j = 0; j < 2; ++j) {
            int n = nbase + wn * 32 + j * 16 + nlo;
            float bias = b1[(size_t)e * HDIM + n];
#pragma unroll
            for (int v = 0; v < 8; ++v) {
                int gr = rbase + wm * 64 + i * 16 + v + rhi;
                float val = acc[i][j][v] + bias;
                val = val > 0.f ? val : 0.f;
                h[((size_t)e * CAP + gr) * HDIM + n] = (__bf16)val;
            }
        }
}

// GEMM2 + fused combine: out[token] = gate * (h @ w2 + b2)
__global__ __launch_bounds__(256) void gemm2_kernel(const __bf16* __restrict__ h,
                                                    const float* __restrict__ w2,
                                                    const float* __restrict__ b2,
                                                    const int* __restrict__ slot_token,
                                                    const float* __restrict__ slot_gate,
                                                    float* __restrict__ out) {
    __shared__ __bf16 As[2 * TILE_ELEMS];
    __shared__ __bf16 Bs[2 * TILE_ELEMS];
    int e = blockIdx.z;
    int rbase = blockIdx.y * 128, nbase = blockIdx.x * 128;
    v8f acc[4][2];
#pragma unroll
    for (int i = 0; i < 4; ++i)
#pragma unroll
        for (int j = 0; j < 2; ++j)
#pragma unroll
            for (int v = 0; v < 8; ++v) acc[i][j][v] = 0.f;

    gemm_core(h + (size_t)e * CAP * HDIM, HDIM,
              w2 + (size_t)e * HDIM * MDIM, MDIM,
              HDIM, rbase, nbase, As, Bs, acc);

    int lane = threadIdx.x & 31, w = threadIdx.x >> 5;
    int wm = w & 1, wn = w >> 1;
    int nlo = lane & 15, rhi = (lane & 16) >> 1;
#pragma unroll
    for (int i = 0; i < 4; ++i)
#pragma unroll
        for (int j = 0; j < 2; ++j) {
            int n = nbase + wn * 32 + j * 16 + nlo;
            float bias = b2[(size_t)e * MDIM + n];
#pragma unroll
            for (int v = 0; v < 8; ++v) {
                int gr   = rbase + wm * 64 + i * 16 + v + rhi;
                int slot = e * CAP + gr;
                int tok  = slot_token[slot];
                if (tok >= 0) {
                    float g = slot_gate[slot];
                    out[(size_t)tok * MDIM + n] = (acc[i][j][v] + bias) * g;
                }
            }
        }
}

// =====================================================================
// host launcher
// =====================================================================
extern "C" void kernel_launch(void* const* d_in, const int* in_sizes, int n_in,
                              void* d_out, int out_size, void* d_ws, size_t ws_size,
                              hipStream_t stream) {
    (void)in_sizes; (void)n_in; (void)out_size; (void)ws_size;
    const float* x  = (const float*)d_in[0];
    const float* wg = (const float*)d_in[1];
    const float* w1 = (const float*)d_in[2];
    const float* b1 = (const float*)d_in[3];
    const float* w2 = (const float*)d_in[4];
    const float* b2 = (const float*)d_in[5];
    float* out = (float*)d_out;

    char* ws = (char*)d_ws;
    int*    idxp       = (int*)  (ws + 0);          // 8192 int
    float*  gate1      = (float*)(ws + 32768);      // 8192 f32
    float*  gates      = (float*)(ws + 65536);      // 8192*8 f32
    float*  me         = (float*)(ws + 327680);     // 8 f32
    float*  cef        = (float*)(ws + 327936);     // 8 f32
    int*    hist       = (int*)  (ws + 328192);     // 32*8
    int*    base       = (int*)  (ws + 329216);     // 32*8
    int*    slot_token = (int*)  (ws + 330240);     // 8192
    float*  slot_gate  = (float*)(ws + 363008);     // 8192
    __bf16* disp       = (__bf16*)(ws + 395776);    // 8*1024*2048 bf16 (32 MB)
    __bf16* hbuf       = (__bf16*)(ws + 33950208ull); // 8*1024*8192 bf16 (128 MB)

    init_kernel   <<<32,    256, 0, stream>>>(slot_token, slot_gate);
    zero_out_kernel<<<16384, 256, 0, stream>>>(out);
    gate_kernel   <<<1024,  256, 0, stream>>>(x, wg, idxp, gate1, gates);
    me_reduce_kernel<<<8,   256, 0, stream>>>(gates, me);
    hist_kernel   <<<32,    256, 0, stream>>>(idxp, hist);
    scan_kernel   <<<1,      32, 0, stream>>>(hist, base, cef);
    assign_kernel <<<32,    256, 0, stream>>>(idxp, gate1, base, slot_token, slot_gate);
    laux_kernel   <<<1,      32, 0, stream>>>(me, cef, out);
    dispatch_kernel<<<8192, 256, 0, stream>>>(x, slot_token, disp);

    dim3 g1(HDIM / 128, CAP / 128, EXP_N);   // (64, 8, 8)
    gemm1_kernel<<<g1, 256, 0, stream>>>(disp, w1, b1, hbuf);

    dim3 g2(MDIM / 128, CAP / 128, EXP_N);   // (16, 8, 8)
    gemm2_kernel<<<g2, 256, 0, stream>>>(hbuf, w2, b2, slot_token, slot_gate, out);
}
